// RSSM_52097953300897
// MI455X (gfx1250) — matrix operand order repaired and checked by reading, not compiled, not run on hardware
//
#include <hip/hip_runtime.h>

// Problem dims (match reference)
#define B_   512
#define T_   64            // Tp1
#define E_   1536
#define A_   12
#define D_   2048
#define S_   64
#define H_   1024
#define OUTW (D_ + 5*S_)   // 2368
#define GATES (3*D_)       // 6144
#define KSA  (S_ + A_)     // 76
#define XAP  96            // padded K of [stoch|act] (multiple of 32)
#define DE_K (D_ + E_)     // 3584

// GEMM tiling: block 256 thr = 8 wave32, tile 128(M) x 64(N) x 32(K)
#define BM 128
#define BN 64
#define BK 32

typedef __bf16 bf16x16 __attribute__((ext_vector_type(16)));
typedef float  f32x8   __attribute__((ext_vector_type(8)));

enum { EP_F32 = 0, EP_ELU_BF16 = 1 };

__device__ __forceinline__ unsigned short f32_to_bf16(float f) {
  union { float f; unsigned u; } x; x.f = f;
  unsigned u = x.u;
  u += 0x7FFFu + ((u >> 16) & 1u);           // round-to-nearest-even
  return (unsigned short)(u >> 16);
}
__device__ __forceinline__ float softplus_(float x) {
  return fmaxf(x, 0.0f) + log1pf(expf(-fabsf(x)));
}
__device__ __forceinline__ float sigmoid_(float x) {
  return 1.0f / (1.0f + expf(-x));
}

// ---------------------------------------------------------------------------
// 16-byte global -> LDS copies. Prefer CDNA5 async-to-LDS (ASYNCcnt) when the
// toolchain exposes the gfx1250 builtins; fall back to VGPR round-trip.
// Builtin signature (from hipcc diagnostic): (int4 AS1*, int4 AS3*, Ii, Ii).
// ---------------------------------------------------------------------------
#if defined(__has_builtin)
#if __has_builtin(__builtin_amdgcn_global_load_async_to_lds_b128) && \
    __has_builtin(__builtin_amdgcn_s_wait_asynccnt)
#define HAVE_ASYNC_LDS 1
#endif
#endif

#if defined(HAVE_ASYNC_LDS)
typedef int v4i_ __attribute__((vector_size(16)));
typedef __attribute__((address_space(1))) v4i_ g1_v4i;
typedef __attribute__((address_space(3))) v4i_ l3_v4i;
__device__ __forceinline__ void copy16_g2l(const void* g, void* l) {
  __builtin_amdgcn_global_load_async_to_lds_b128((g1_v4i*)g, (l3_v4i*)l, 0, 0);
}
__device__ __forceinline__ void copy16_join() {  // wave's async copies complete
  __builtin_amdgcn_s_wait_asynccnt(0);
}
#else
__device__ __forceinline__ void copy16_g2l(const void* g, void* l) {
  *(uint4*)l = *(const uint4*)g;
}
__device__ __forceinline__ void copy16_join() {}
#endif

// ---------------------------------------------------------------------------
// All-bf16 tiled WMMA GEMM:  C[m,n] = epi( sum_k A[m,k] * W[n,k] + bias[n] )
// A: bf16 [M,K] stride lda ; W: bf16 [N,K] stride ldw (both 16B-row-aligned).
// REQUIRES M % 128 == 0, N % 64 == 0, K % 32 == 0 (guaranteed by caller).
// Ping-pong LDS; copies for K-tile k+1 run async under the WMMAs of tile k.
// Wave grid 4(M) x 2(N); wave tile 32x32 -> 4 v_wmma_f32_16x16x32_bf16 / step.
// ---------------------------------------------------------------------------
__global__ __launch_bounds__(256) void gemm_wmma_bf16(
    const unsigned short* __restrict__ A, int lda,
    const unsigned short* __restrict__ W, int ldw,
    const float* __restrict__ bias,
    void* __restrict__ Cv, long long ldc,
    int M, int N, int K, int epilogue)
{
  // rows padded to 40 shorts (80B, 16B aligned) to dodge LDS bank conflicts
  __shared__ alignas(16) unsigned short sA[2][BM][40];
  __shared__ alignas(16) unsigned short sB[2][BN][40];

  const int tid   = threadIdx.x;
  const int lane  = tid & 31;
  const int wave  = tid >> 5;       // 0..7
  const int wm    = wave & 3;       // wave position in M (4)
  const int wn    = wave >> 2;      // wave position in N (2)
  const int m0    = blockIdx.y * BM;
  const int n0    = blockIdx.x * BN;
  const int khalf = lane >> 4;      // K-half select per ISA 16-bit layout
  const int l16   = lane & 15;

  // copy slots: 16B = 8 bf16.  A tile 128x32 = 512 slots (2/thread);
  // B tile 64x32 = 256 slots (1/thread).
  const int rA0 = tid >> 2;               // rows 0..63   (slot set 0)
  const int rA1 = (tid >> 2) + 64;        // rows 64..127 (slot set 1)
  const int kb  = (tid & 3) * 8;          // 0,8,16,24

  f32x8 acc[2][2] = {};
  const int kSteps = K / BK;

  // prologue: issue K-tile 0 into buffer 0
  copy16_g2l(&A[(long long)(m0 + rA0) * lda + kb], &sA[0][rA0][kb]);
  copy16_g2l(&A[(long long)(m0 + rA1) * lda + kb], &sA[0][rA1][kb]);
  copy16_g2l(&W[(long long)(n0 + rA0) * ldw + kb], &sB[0][rA0][kb]);

  for (int ks = 0; ks < kSteps; ++ks) {
    const int buf = ks & 1;
    copy16_join();        // this wave's copies for tile ks have landed in LDS
    __syncthreads();      // all waves' copies done; all reads of buf^1 done

    if (ks + 1 < kSteps) {  // issue tile ks+1 into the other buffer (overlaps WMMA)
      const int k1 = (ks + 1) * BK;
      copy16_g2l(&A[(long long)(m0 + rA0) * lda + k1 + kb], &sA[buf ^ 1][rA0][kb]);
      copy16_g2l(&A[(long long)(m0 + rA1) * lda + k1 + kb], &sA[buf ^ 1][rA1][kb]);
      copy16_g2l(&W[(long long)(n0 + rA0) * ldw + k1 + kb], &sB[buf ^ 1][rA0][kb]);
    }

    // fragments per CDNA5 16-bit WMMA VGPR layout
    union FA { bf16x16 v; uint4 q[2]; } fa[2];
    #pragma unroll
    for (int mt = 0; mt < 2; ++mt) {
      const uint4* p = (const uint4*)&sA[buf][wm * 32 + mt * 16 + l16][khalf * 8];
      fa[mt].q[0] = p[0];
      fa[mt].q[1] = p[2];       // +32B => K += 16
    }
    union FB { bf16x16 v; uint4 q[2]; } fb[2];
    #pragma unroll
    for (int nt = 0; nt < 2; ++nt) {
      const uint4* p = (const uint4*)&sB[buf][wn * 32 + nt * 16 + l16][khalf * 8];
      fb[nt].q[0] = p[0];
      fb[nt].q[1] = p[2];
    }

    #pragma unroll
    for (int mt = 0; mt < 2; ++mt)
      #pragma unroll
      for (int nt = 0; nt < 2; ++nt)
        acc[mt][nt] = __builtin_amdgcn_wmma_f32_16x16x32_bf16(
            false, fa[mt].v, false, fb[nt].v, (short)0, acc[mt][nt], false, false);
  }

  // epilogue + store: C 16x16 f32 layout (VGPR r <-> M = 8*khalf + r, l16 <-> N)
  #pragma unroll
  for (int mt = 0; mt < 2; ++mt) {
    const int mBase = m0 + wm * 32 + mt * 16 + khalf * 8;
    #pragma unroll
    for (int nt = 0; nt < 2; ++nt) {
      const int n = n0 + wn * 32 + nt * 16 + l16;
      const float bv = bias[n];
      #pragma unroll
      for (int r = 0; r < 8; ++r) {
        float v = acc[mt][nt][r] + bv;
        if (epilogue == EP_ELU_BF16) {
          v = (v > 0.0f) ? v : expm1f(v);
          ((unsigned short*)Cv)[(long long)(mBase + r) * ldc + n] = f32_to_bf16(v);
        } else {
          ((float*)Cv)[(long long)(mBase + r) * ldc + n] = v;
        }
      }
    }
  }
}

// ---------------------------------------------------------------------------
// Weight conversion: f32 [N,Ksrc] -> bf16 [N,Kdst] with zero K-padding
// ---------------------------------------------------------------------------
__global__ void k_cvt_pad(unsigned short* __restrict__ dst, const float* __restrict__ src,
                          int N, int Ksrc, int Kdst) {
  int i = blockIdx.x * blockDim.x + threadIdx.x;
  if (i >= N * Kdst) return;
  int n = i / Kdst, k = i % Kdst;
  dst[i] = (k < Ksrc) ? f32_to_bf16(src[(long long)n * Ksrc + k]) : (unsigned short)0;
}

// ---------------------------------------------------------------------------
// Elementwise kernels
// ---------------------------------------------------------------------------
__global__ void k_init(float* deter, unsigned short* xa) {
  int i = blockIdx.x * blockDim.x + threadIdx.x;
  if (i < B_ * D_)  deter[i] = 0.0f;
  if (i < B_ * XAP) xa[i]    = 0;      // bf16 zero (also zeroes K padding)
}

__global__ void k_build_act(unsigned short* xa, const float* __restrict__ action, int t) {
  int i = blockIdx.x * blockDim.x + threadIdx.x;
  if (i >= B_ * A_) return;
  int b = i / A_, c = i % A_;
  float v = (t == 0) ? 0.0f : action[((long long)b * (T_ - 1) + (t - 1)) * A_ + c];
  xa[b * XAP + S_ + c] = f32_to_bf16(v);
}

__global__ void k_gru(const float* __restrict__ gi, const float* __restrict__ gh,
                      float* deter, unsigned short* deter_bf, float* out, int t) {
  int i = blockIdx.x * blockDim.x + threadIdx.x;
  if (i >= B_ * D_) return;
  int b = i / D_, d = i % D_;
  const float* gib = gi + (long long)b * GATES;
  const float* ghb = gh + (long long)b * GATES;
  float r = sigmoid_(gib[d]        + ghb[d]);
  float z = sigmoid_(gib[D_ + d]   + ghb[D_ + d]);
  float n = tanhf  (gib[2*D_ + d] + r * ghb[2*D_ + d]);
  float h = (1.0f - z) * n + z * deter[i];
  deter[i]    = h;
  deter_bf[i] = f32_to_bf16(h);
  out[((long long)b * T_ + t) * OUTW + d] = h;
}

__global__ void k_de(unsigned short* de, const unsigned short* __restrict__ deter_bf,
                     const float* __restrict__ embed, int t) {
  int i = blockIdx.x * blockDim.x + threadIdx.x;
  if (i >= B_ * DE_K) return;
  int b = i / DE_K, c = i % DE_K;
  de[i] = (c < D_) ? deter_bf[b * D_ + c]
                   : f32_to_bf16(embed[((long long)b * T_ + t) * E_ + (c - D_)]);
}

__global__ void k_heads(float* out, const float* __restrict__ noise,
                        unsigned short* xa, int t) {
  int i = blockIdx.x * blockDim.x + threadIdx.x;
  if (i >= B_ * S_) return;
  int b = i / S_, s = i % S_;
  long long base = ((long long)b * T_ + t) * OUTW;
  float psr = out[base + D_ + 2*S_ + s];
  out[base + D_ + 2*S_ + s] = softplus_(psr) + 0.1f;
  float qm  = out[base + D_ + 3*S_ + s];
  float qsr = out[base + D_ + 4*S_ + s];
  float qs  = softplus_(qsr) + 0.1f;
  out[base + D_ + 4*S_ + s] = qs;
  float eps = noise[((long long)b * T_ + t) * S_ + s];
  float st  = qm + qs * eps;
  out[base + D_ + s] = st;                 // stoch slice
  xa[b * XAP + s]    = f32_to_bf16(st);    // feeds next step
}

// ---------------------------------------------------------------------------
extern "C" void kernel_launch(void* const* d_in, const int* in_sizes, int n_in,
                              void* d_out, int out_size, void* d_ws, size_t ws_size,
                              hipStream_t stream) {
  (void)in_sizes; (void)n_in; (void)out_size; (void)ws_size;
  const float* embed  = (const float*)d_in[0];
  const float* action = (const float*)d_in[1];
  const float* noise  = (const float*)d_in[2];
  const float* W_sa = (const float*)d_in[3];  const float* b_sa = (const float*)d_in[4];
  const float* W_ih = (const float*)d_in[5];  const float* b_ih = (const float*)d_in[6];
  const float* W_hh = (const float*)d_in[7];  const float* b_hh = (const float*)d_in[8];
  const float* W_pr = (const float*)d_in[9];  const float* b_pr = (const float*)d_in[10];
  const float* W_pm = (const float*)d_in[11]; const float* b_pm = (const float*)d_in[12];
  const float* W_ps = (const float*)d_in[13]; const float* b_ps = (const float*)d_in[14];
  const float* W_po = (const float*)d_in[15]; const float* b_po = (const float*)d_in[16];
  const float* W_qm = (const float*)d_in[17]; const float* b_qm = (const float*)d_in[18];
  const float* W_qs = (const float*)d_in[19]; const float* b_qs = (const float*)d_in[20];
  float* out = (float*)d_out;

  // ---- workspace partition (f32 region then bf16 region), ~88 MB total ----
  float* wsf   = (float*)d_ws;
  float* gi    = wsf;                         // [512,6144] f32
  float* gh    = gi    + (size_t)B_ * GATES;  // [512,6144] f32
  float* deter = gh    + (size_t)B_ * GATES;  // [512,2048] f32 state
  unsigned short* wsb = (unsigned short*)(deter + (size_t)B_ * D_);
  unsigned short* x_bf  = wsb;                          // [512,1024]
  unsigned short* dt_bf = x_bf  + (size_t)B_ * H_;      // [512,2048]
  unsigned short* xa_bf = dt_bf + (size_t)B_ * D_;      // [512,96]
  unsigned short* de_bf = xa_bf + (size_t)B_ * XAP;     // [512,3584]
  unsigned short* hp_bf = de_bf + (size_t)B_ * DE_K;    // [512,1024]
  unsigned short* hq_bf = hp_bf + (size_t)B_ * H_;      // [512,1024]
  unsigned short* w_sa  = hq_bf + (size_t)B_ * H_;      // [1024,96] (padded)
  unsigned short* w_ih  = w_sa  + (size_t)H_ * XAP;     // [6144,1024]
  unsigned short* w_hh  = w_ih  + (size_t)GATES * H_;   // [6144,2048]
  unsigned short* w_pr  = w_hh  + (size_t)GATES * D_;   // [1024,2048]
  unsigned short* w_po  = w_pr  + (size_t)H_ * D_;      // [1024,3584]
  unsigned short* w_pm  = w_po  + (size_t)H_ * DE_K;    // [64,1024]
  unsigned short* w_ps  = w_pm  + (size_t)S_ * H_;
  unsigned short* w_qm  = w_ps  + (size_t)S_ * H_;
  unsigned short* w_qs  = w_qm  + (size_t)S_ * H_;

  const long long ldcOut = (long long)T_ * OUTW;
  const dim3 blk(256);
  const dim3 gN1024(H_ / BN,    B_ / BM);
  const dim3 gGates(GATES / BN, B_ / BM);
  const dim3 gHead (S_ / BN,    B_ / BM);
  auto cvb = [](size_t n) { return dim3((unsigned)((n + 255) / 256)); };

  // one-time (per launch) weight conversion to bf16; L2-resident thereafter
  k_cvt_pad<<<cvb((size_t)H_ * XAP),    blk, 0, stream>>>(w_sa, W_sa, H_,    KSA,  XAP);
  k_cvt_pad<<<cvb((size_t)GATES * H_),  blk, 0, stream>>>(w_ih, W_ih, GATES, H_,   H_);
  k_cvt_pad<<<cvb((size_t)GATES * D_),  blk, 0, stream>>>(w_hh, W_hh, GATES, D_,   D_);
  k_cvt_pad<<<cvb((size_t)H_ * D_),     blk, 0, stream>>>(w_pr, W_pr, H_,    D_,   D_);
  k_cvt_pad<<<cvb((size_t)H_ * DE_K),   blk, 0, stream>>>(w_po, W_po, H_,    DE_K, DE_K);
  k_cvt_pad<<<cvb((size_t)S_ * H_),     blk, 0, stream>>>(w_pm, W_pm, S_,    H_,   H_);
  k_cvt_pad<<<cvb((size_t)S_ * H_),     blk, 0, stream>>>(w_ps, W_ps, S_,    H_,   H_);
  k_cvt_pad<<<cvb((size_t)S_ * H_),     blk, 0, stream>>>(w_qm, W_qm, S_,    H_,   H_);
  k_cvt_pad<<<cvb((size_t)S_ * H_),     blk, 0, stream>>>(w_qs, W_qs, S_,    H_,   H_);

  k_init<<<cvb((size_t)B_ * D_), blk, 0, stream>>>(deter, xa_bf);

  for (int t = 0; t < T_; ++t) {
    k_build_act<<<cvb((size_t)B_ * A_), blk, 0, stream>>>(xa_bf, action, t);

    // x = ELU([stoch|act] @ W_sa^T + b_sa)   K=96(padded) -> bf16
    gemm_wmma_bf16<<<gN1024, blk, 0, stream>>>(
        xa_bf, XAP, w_sa, XAP, b_sa, x_bf, H_, B_, H_, XAP, EP_ELU_BF16);

    // gi = x @ W_ih^T + b_ih   (f32 for GRU)
    gemm_wmma_bf16<<<gGates, blk, 0, stream>>>(
        x_bf, H_, w_ih, H_, b_ih, gi, GATES, B_, GATES, H_, EP_F32);

    // gh = deter @ W_hh^T + b_hh
    gemm_wmma_bf16<<<gGates, blk, 0, stream>>>(
        dt_bf, D_, w_hh, D_, b_hh, gh, GATES, B_, GATES, D_, EP_F32);

    // GRU update -> deter (f32 + bf16) + out[:,t,0:2048]
    k_gru<<<cvb((size_t)B_ * D_), blk, 0, stream>>>(gi, gh, deter, dt_bf, out, t);

    // de = [deter | embed_t]  (bf16)
    k_de<<<cvb((size_t)B_ * DE_K), blk, 0, stream>>>(de_bf, dt_bf, embed, t);

    // hp = ELU(deter @ W_pr^T + b_pr) -> bf16
    gemm_wmma_bf16<<<gN1024, blk, 0, stream>>>(
        dt_bf, D_, w_pr, D_, b_pr, hp_bf, H_, B_, H_, D_, EP_ELU_BF16);

    // hq = ELU([deter|e] @ W_po^T + b_po) -> bf16
    gemm_wmma_bf16<<<gN1024, blk, 0, stream>>>(
        de_bf, DE_K, w_po, DE_K, b_po, hq_bf, H_, B_, H_, DE_K, EP_ELU_BF16);

    // heads straight into strided output slices (f32), ldc = 64*2368
    gemm_wmma_bf16<<<gHead, blk, 0, stream>>>(   // pm
        hp_bf, H_, w_pm, H_, b_pm, out + (long long)t * OUTW + D_ + S_,   ldcOut, B_, S_, H_, EP_F32);
    gemm_wmma_bf16<<<gHead, blk, 0, stream>>>(   // ps (raw)
        hp_bf, H_, w_ps, H_, b_ps, out + (long long)t * OUTW + D_ + 2*S_, ldcOut, B_, S_, H_, EP_F32);
    gemm_wmma_bf16<<<gHead, blk, 0, stream>>>(   // qm
        hq_bf, H_, w_qm, H_, b_qm, out + (long long)t * OUTW + D_ + 3*S_, ldcOut, B_, S_, H_, EP_F32);
    gemm_wmma_bf16<<<gHead, blk, 0, stream>>>(   // qs (raw)
        hq_bf, H_, w_qs, H_, b_qs, out + (long long)t * OUTW + D_ + 4*S_, ldcOut, B_, S_, H_, EP_F32);

    // softplus heads, reparameterized sample, feed stoch to next step
    k_heads<<<cvb((size_t)B_ * S_), blk, 0, stream>>>(out, noise, xa_bf, t);
  }
}
